// Decoder_69630009803359
// MI455X (gfx1250) — matrix operand order repaired and checked by reading, not compiled, not run on hardware
//
#include <hip/hip_runtime.h>
#include <math.h>

// ---------------------------------------------------------------------------
// Types for CDNA5 WMMA
// ---------------------------------------------------------------------------
typedef __attribute__((ext_vector_type(16))) _Float16 v16h;
typedef __attribute__((ext_vector_type(8)))  _Float16 v8h;
typedef __attribute__((ext_vector_type(8)))  float    v8f;

__device__ __forceinline__ v16h cat8(v8h lo, v8h hi) {
    return __builtin_shufflevector(lo, hi, 0,1,2,3,4,5,6,7,8,9,10,11,12,13,14,15);
}

__device__ __forceinline__ float sigmoidf_(float x) { return 1.0f / (1.0f + expf(-x)); }
__device__ __forceinline__ float softplusf_(float x) { return (x > 20.0f) ? x : logf(1.0f + expf(x)); }

// Model constants (match reference)
#define BSZ   32
#define TIN   200
#define DENC  512
#define HH    1024
#define KMIX  5
#define NOUT  80
#define MAXR  2
#define TMEL  800
#define PRE_N 256
#define NSTEP 400          // TMEL / r
#define HQ    102          // H // 10
#define KA    1792         // 768 + 1024  (attn LSTM concat K)
#define KD    2048         // 1024 + 1024 (dec LSTM concat K)
#define KP    1536         // H + DENC    (proj K)
#define NP    176          // 160 proj + 2 stop, padded to 16
#define LPAD  804          // 800 + 2*halo

// ---------------------------------------------------------------------------
// Generic WMMA GEMM:  out(32,N) = X(32,K) * W(N,K)^T + bias
// One wave per 16x16 tile; 2 M-tiles, N/16 N-tiles.
// ---------------------------------------------------------------------------
__global__ void __launch_bounds__(256)
gemm_xw_kernel(const _Float16* __restrict__ X, const _Float16* __restrict__ W,
               const float* __restrict__ bias, float* __restrict__ out,
               int K, int N, int nJobs) {
    int wave = (blockIdx.x * blockDim.x + threadIdx.x) >> 5;
    if (wave >= nJobs) return;                 // wave-uniform: EXEC stays all-ones
    int lane  = threadIdx.x & 31;
    int mtile = wave & 1;
    int ntile = wave >> 1;
    int mrow  = mtile * 16 + (lane & 15);
    int ncol  = ntile * 16 + (lane & 15);
    int koffA = (lane & 16) ? 8 : 0;           // ISA A-frag layout
    int koffB = (lane & 16) ? 8 : 0;           // ISA B-frag layout
    const _Float16* xbase = X + (size_t)mrow * K + koffA;
    const _Float16* wbase = W + (size_t)ncol * K + koffB;
    v8f acc = {};
    for (int k0 = 0; k0 < K; k0 += 32) {
        __builtin_prefetch(wbase + k0 + 128, 0, 1);
        v8h a0 = *(const v8h*)(xbase + k0);
        v8h a1 = *(const v8h*)(xbase + k0 + 16);
        v8h b0 = *(const v8h*)(wbase + k0);
        v8h b1 = *(const v8h*)(wbase + k0 + 16);
        acc = __builtin_amdgcn_wmma_f32_16x16x32_f16(
                false, cat8(a0, a1), false, cat8(b0, b1),
                (short)0, acc, false, false);
    }
    int mb = mtile * 16 + ((lane & 16) ? 8 : 0);
    float bv = bias ? bias[ncol] : 0.0f;
#pragma unroll
    for (int v = 0; v < 8; ++v)
        out[(size_t)(mb + v) * N + ncol] = acc[v] + bv;
}

// ---------------------------------------------------------------------------
// LSTM cell + zoneout (eval blend).  gates layout (32,4096) = [i f g o] blocks.
// Writes fp32 h,c and f16 copies of new h into up to two x-concat buffers.
// ---------------------------------------------------------------------------
__global__ void lstm_cell_kernel(const float* __restrict__ gates,
                                 float* __restrict__ h, float* __restrict__ c,
                                 _Float16* __restrict__ d0, int ld0, int off0,
                                 _Float16* __restrict__ d1, int ld1, int off1) {
    int idx = blockIdx.x * blockDim.x + threadIdx.x;
    if (idx >= BSZ * HH) return;
    int b = idx >> 10, n = idx & 1023;
    const float* g = gates + (size_t)b * 4096;
    float gi = g[n], gf = g[1024 + n], gg = g[2048 + n], go = g[3072 + n];
    float cold = c[idx], hold = h[idx];
    float cn = sigmoidf_(gf) * cold + sigmoidf_(gi) * tanhf(gg);
    float hn = sigmoidf_(go) * tanhf(cn);
    float hz = 0.1f * hold + 0.9f * hn;
    float cz = 0.1f * cold + 0.9f * cn;
    h[idx] = hz; c[idx] = cz;
    _Float16 hh16 = (_Float16)hz;
    if (d0) d0[(size_t)b * ld0 + off0 + n] = hh16;
    if (d1) d1[(size_t)b * ld1 + off1 + n] = hh16;
}

// ---------------------------------------------------------------------------
// Per-step: prenet + GMM attention + context.  One block per batch element.
// ---------------------------------------------------------------------------
__global__ void __launch_bounds__(256)
attn_step_kernel(int step,
                 const float* __restrict__ mels, const float* __restrict__ enc,
                 const float* __restrict__ pw1, const float* __restrict__ pb1,
                 const float* __restrict__ pw2, const float* __restrict__ pb2,
                 const float* __restrict__ aw1, const float* __restrict__ ab1,
                 const float* __restrict__ aw2, const float* __restrict__ ab2,
                 const float* __restrict__ ah, float* __restrict__ mu,
                 _Float16* __restrict__ xcatA, _Float16* __restrict__ xcatP,
                 float* __restrict__ attn_out) {
    __shared__ float prev[NOUT];
    __shared__ float p1[PRE_N];
    __shared__ float p2[PRE_N];
    __shared__ float t1[HQ + 2];
    __shared__ float gbk[3 * KMIX];
    __shared__ float coef[KMIX], muk[KMIX], sk[KMIX];
    __shared__ float alpha[TIN];
    int b = blockIdx.x, tid = threadIdx.x;

    if (tid < NOUT) {
        float v = 0.0f;
        if (step > 0) v = mels[((size_t)b * TMEL + (2 * step - 1)) * NOUT + tid];
        prev[tid] = v;
    }
    __syncthreads();
    { // prenet layer 1: 80 -> 256, relu
        float s = pb1[tid];
        const float* w = pw1 + (size_t)tid * NOUT;
        for (int j = 0; j < NOUT; ++j) s += prev[j] * w[j];
        p1[tid] = fmaxf(s, 0.0f);
    }
    __syncthreads();
    { // prenet layer 2: 256 -> 256, relu
        float s = pb2[tid];
        const float* w = pw2 + (size_t)tid * PRE_N;
        for (int j = 0; j < PRE_N; ++j) s += p1[j] * w[j];
        p2[tid] = fmaxf(s, 0.0f);
    }
    // attention MLP: tanh(ah @ W1^T + b1)
    if (tid < HQ) {
        float s = ab1[tid];
        const float* w  = aw1 + (size_t)tid * HH;
        const float* hb = ah + (size_t)b * HH;
        for (int j = 0; j < HH; ++j) s += hb[j] * w[j];
        t1[tid] = tanhf(s);
    }
    __syncthreads();
    if (tid < 3 * KMIX) {
        float s = ab2[tid];
        const float* w = aw2 + (size_t)tid * HQ;
        for (int q = 0; q < HQ; ++q) s += t1[q] * w[q];
        gbk[tid] = s;
    }
    __syncthreads();
    if (tid < KMIX) { // GMM parameters
        float mx = gbk[0];
        for (int k = 1; k < KMIX; ++k) mx = fmaxf(mx, gbk[k]);
        float den = 0.0f;
        for (int k = 0; k < KMIX; ++k) den += expf(gbk[k] - mx);
        float g   = expf(gbk[tid] - mx) / den + 1e-5f;
        float sig = softplusf_(gbk[KMIX + tid]) + 1e-5f;
        float m   = mu[b * KMIX + tid] + softplusf_(gbk[2 * KMIX + tid]);
        mu[b * KMIX + tid] = m;
        float z = sqrtf(6.283185307179586f * sig * sig) + 1e-5f;
        coef[tid] = g / z; muk[tid] = m; sk[tid] = sig;
    }
    __syncthreads();
    if (tid < TIN) { // alpha over grid
        float j = (float)tid;
        float a = 0.0f;
        for (int k = 0; k < KMIX; ++k) {
            float d = (muk[k] - j) / sk[k];
            a += coef[k] * expf(-0.5f * d * d);
        }
        alpha[tid] = a;
        attn_out[((size_t)b * NSTEP + step) * TIN + tid] = a;
    }
    __syncthreads();
    // context: ctx[d] = sum_t alpha[t]*enc[b,t,d]
    for (int d = tid; d < DENC; d += 256) {
        float s = 0.0f;
        const float* e = enc + (size_t)b * TIN * DENC + d;
        for (int t = 0; t < TIN; ++t) s += alpha[t] * e[(size_t)t * DENC];
        _Float16 hv = (_Float16)s;
        xcatA[(size_t)b * KA + d] = hv;             // attn LSTM input [ctx|pre|h]
        xcatP[(size_t)b * KP + HH + d] = hv;        // proj input [dh|ctx]
    }
    xcatA[(size_t)b * KA + DENC + tid] = (_Float16)p2[tid];
}

// ---------------------------------------------------------------------------
// Scatter projection results into output regions 2 (mels) and 3 (stop)
// ---------------------------------------------------------------------------
__global__ void write_proj_kernel(const float* __restrict__ projout,
                                  float* __restrict__ r2, float* __restrict__ r3,
                                  int step) {
    int b = blockIdx.x, n = threadIdx.x;
    if (n < NOUT * MAXR) {
        int o = n >> 1, ri = n & 1;
        r2[((size_t)b * TMEL + 2 * step + ri) * NOUT + o] = projout[b * NP + n];
    } else if (n < NOUT * MAXR + MAXR) {
        r3[(size_t)b * TMEL + 2 * step + (n - NOUT * MAXR)] = projout[b * NP + n];
    }
}

// ---------------------------------------------------------------------------
// Postnet conv layer as implicit-GEMM WMMA.
// in:  (B, 804, Cpin) f16 with zero halo;  wgt: (Cout, 5, Cpin) f16
// mid layers -> f16 act buffer; last layer -> fp32 residual-added output.
// ---------------------------------------------------------------------------
__global__ void __launch_bounds__(256)
post_conv_kernel(const _Float16* __restrict__ in, const _Float16* __restrict__ wgt,
                 const float* __restrict__ gscale, const float* __restrict__ bterm,
                 _Float16* __restrict__ outh, float* __restrict__ outf,
                 const float* __restrict__ resid, int Cpin, int Cout, int nJobs) {
    int wave = (blockIdx.x * blockDim.x + threadIdx.x) >> 5;
    if (wave >= nJobs) return;
    int lane = threadIdx.x & 31;
    int ntiles = Cout >> 4;
    int ntile = wave % ntiles;
    int rem   = wave / ntiles;
    int ltile = rem % 50;
    int b     = rem / 50;
    int lm  = ltile * 16 + (lane & 15);
    int co  = ntile * 16 + (lane & 15);
    int koffA = (lane & 16) ? 8 : 0;
    int koffB = (lane & 16) ? 8 : 0;
    v8f acc = {};
    for (int dk = 0; dk < 5; ++dk) {
        const _Float16* abase = in  + ((size_t)(b * LPAD) + lm + dk) * Cpin + koffA;
        const _Float16* wbase = wgt + ((size_t)co * 5 + dk) * Cpin + koffB;
        for (int kc = 0; kc < Cpin; kc += 32) {
            v8h a0 = *(const v8h*)(abase + kc);
            v8h a1 = *(const v8h*)(abase + kc + 16);
            v8h b0 = *(const v8h*)(wbase + kc);
            v8h b1 = *(const v8h*)(wbase + kc + 16);
            acc = __builtin_amdgcn_wmma_f32_16x16x32_f16(
                    false, cat8(a0, a1), false, cat8(b0, b1),
                    (short)0, acc, false, false);
        }
    }
    float sc = gscale[co] * rsqrtf(1.0f + 1e-5f);
    float bb = bterm[co];
    int lb = ltile * 16 + ((lane & 16) ? 8 : 0);
#pragma unroll
    for (int v = 0; v < 8; ++v) {
        float y = tanhf(acc[v] * sc + bb);
        int l = lb + v;
        if (outh) {
            outh[((size_t)(b * LPAD) + 2 + l) * Cout + co] = (_Float16)y;
        } else {
            size_t o = ((size_t)(b * TMEL) + l) * NOUT + co;
            outf[o] = resid[o] + y;
        }
    }
}

// ---------------------------------------------------------------------------
// Weight packing / init kernels (run once per call)
// ---------------------------------------------------------------------------
__global__ void concat2h_kernel(_Float16* __restrict__ dst, const float* __restrict__ a,
                                const float* __restrict__ b, int Ka, int Kb, int total) {
    int idx = blockIdx.x * blockDim.x + threadIdx.x;
    if (idx >= total) return;
    int K = Ka + Kb;
    int n = idx / K, k = idx % K;
    float v = (k < Ka) ? a[(size_t)n * Ka + k] : b[(size_t)n * Kb + (k - Ka)];
    dst[idx] = (_Float16)v;
}

__global__ void addbias_kernel(float* __restrict__ dst, const float* __restrict__ a,
                               const float* __restrict__ b, int n) {
    int i = blockIdx.x * blockDim.x + threadIdx.x;
    if (i < n) dst[i] = a[i] + b[i];
}

__global__ void prep_projw_kernel(_Float16* __restrict__ dst, const float* __restrict__ pw,
                                  const float* __restrict__ sw, int total) {
    int idx = blockIdx.x * blockDim.x + threadIdx.x;
    if (idx >= total) return;
    int n = idx / KP, k = idx % KP;
    float v = 0.0f;
    if (n < 160)      v = pw[(size_t)n * KP + k];
    else if (n < 162) v = sw[(size_t)(n - 160) * KP + k];
    dst[idx] = (_Float16)v;
}

__global__ void prep_projb_kernel(float* __restrict__ dst, const float* __restrict__ sb) {
    int n = threadIdx.x;
    if (n < NP) dst[n] = (n >= 160 && n < 162) ? sb[n - 160] : 0.0f;
}

__global__ void prep_postw_kernel(_Float16* __restrict__ dst, const float* __restrict__ src,
                                  int Cout, int Cin, int Cpin, int total) {
    int idx = blockIdx.x * blockDim.x + threadIdx.x;
    if (idx >= total) return;
    int ci = idx % Cpin; int rem = idx / Cpin;
    int dk = rem % 5;    int co  = rem / 5;
    float v = (ci < Cin) ? src[((size_t)co * Cin + ci) * 5 + dk] : 0.0f;
    dst[idx] = (_Float16)v;
}

__global__ void post_prep_input_kernel(const float* __restrict__ r2,
                                       _Float16* __restrict__ act, int total) {
    int idx = blockIdx.x * blockDim.x + threadIdx.x;
    if (idx >= total) return;
    int c = idx % 96; int rem = idx / 96;
    int row = rem % LPAD; int b = rem / LPAD;
    float v = 0.0f;
    int l = row - 2;
    if (c < NOUT && l >= 0 && l < TMEL) v = r2[((size_t)b * TMEL + l) * NOUT + c];
    act[idx] = (_Float16)v;
}

__global__ void zero_f_kernel(float* p, int n) {
    int i = blockIdx.x * blockDim.x + threadIdx.x;
    if (i < n) p[i] = 0.0f;
}
__global__ void zero_h_kernel(_Float16* p, int n) {
    int i = blockIdx.x * blockDim.x + threadIdx.x;
    if (i < n) p[i] = (_Float16)0.0f;
}

// ---------------------------------------------------------------------------
// Host launcher
// ---------------------------------------------------------------------------
static inline int nblk(long long n, int t) { return (int)((n + t - 1) / t); }

extern "C" void kernel_launch(void* const* d_in, const int* in_sizes, int n_in,
                              void* d_out, int out_size, void* d_ws, size_t ws_size,
                              hipStream_t stream) {
    (void)in_sizes; (void)n_in; (void)out_size; (void)ws_size;
    const float* enc   = (const float*)d_in[0];
    const float* mels  = (const float*)d_in[1];
    const float* pw1   = (const float*)d_in[2];
    const float* pb1   = (const float*)d_in[3];
    const float* pw2   = (const float*)d_in[4];
    const float* pb2   = (const float*)d_in[5];
    const float* aw1   = (const float*)d_in[6];
    const float* ab1   = (const float*)d_in[7];
    const float* aw2   = (const float*)d_in[8];
    const float* ab2   = (const float*)d_in[9];
    const float* aihw  = (const float*)d_in[10];
    const float* aihb  = (const float*)d_in[11];
    const float* ahhw  = (const float*)d_in[12];
    const float* ahhb  = (const float*)d_in[13];
    const float* dihw  = (const float*)d_in[14];
    const float* dihb  = (const float*)d_in[15];
    const float* dhhw  = (const float*)d_in[16];
    const float* dhhb  = (const float*)d_in[17];
    const float* projw = (const float*)d_in[18];
    const float* stopw = (const float*)d_in[19];
    const float* stopb = (const float*)d_in[20];
    const float* postw[5]; const float* postg[5]; const float* postb[5];
    for (int i = 0; i < 5; ++i) {
        postw[i] = (const float*)d_in[21 + i];
        postg[i] = (const float*)d_in[26 + i];
        postb[i] = (const float*)d_in[31 + i];
    }

    float* outf = (float*)d_out;
    float* r1 = outf;                                    // (B,800,80) postnet
    float* r2 = outf + (size_t)BSZ * TMEL * NOUT;        // (B,800,80) pre-postnet
    float* r3 = r2  + (size_t)BSZ * TMEL * NOUT;         // (B,800)    stop
    float* r4 = r3  + (size_t)BSZ * TMEL;                // (B,400,200) attn

    char* ws = (char*)d_ws;
    size_t cur = 0;
    auto alloc = [&](size_t bytes) -> char* {
        char* p = ws + cur;
        cur = (cur + bytes + 255) & ~(size_t)255;
        return p;
    };
    _Float16* WcatA = (_Float16*)alloc((size_t)4096 * KA * 2);
    _Float16* WcatD = (_Float16*)alloc((size_t)4096 * KD * 2);
    _Float16* Wproj = (_Float16*)alloc((size_t)NP * KP * 2);
    float* bcatA = (float*)alloc(4096 * 4);
    float* bcatD = (float*)alloc(4096 * 4);
    float* bproj = (float*)alloc(NP * 4);
    _Float16* wp[5];
    int cpin[5]  = { 96, 512, 512, 512, 512 };
    int cin[5]   = { 80, 512, 512, 512, 512 };
    int cout[5]  = { 512, 512, 512, 512, 80 };
    for (int i = 0; i < 5; ++i)
        wp[i] = (_Float16*)alloc((size_t)cout[i] * 5 * cpin[i] * 2);
    _Float16* xcatA = (_Float16*)alloc((size_t)BSZ * KA * 2);
    _Float16* xcatD = (_Float16*)alloc((size_t)BSZ * KD * 2);
    _Float16* xcatP = (_Float16*)alloc((size_t)BSZ * KP * 2);
    float* gates   = (float*)alloc((size_t)BSZ * 4096 * 4);
    float* ah = (float*)alloc((size_t)BSZ * HH * 4);
    float* ac = (float*)alloc((size_t)BSZ * HH * 4);
    float* dh = (float*)alloc((size_t)BSZ * HH * 4);
    float* dc = (float*)alloc((size_t)BSZ * HH * 4);
    float* mu = (float*)alloc((size_t)BSZ * KMIX * 4);
    float* projout = (float*)alloc((size_t)BSZ * NP * 4);
    _Float16* actA = (_Float16*)alloc((size_t)BSZ * LPAD * 96 * 2);
    _Float16* actB = (_Float16*)alloc((size_t)BSZ * LPAD * 512 * 2);
    _Float16* actC = (_Float16*)alloc((size_t)BSZ * LPAD * 512 * 2);

    // ---- one-time packing / init ----
    {
        int tA = 4096 * KA, tD = 4096 * KD;
        concat2h_kernel<<<nblk(tA,256),256,0,stream>>>(WcatA, aihw, ahhw, 768, 1024, tA);
        concat2h_kernel<<<nblk(tD,256),256,0,stream>>>(WcatD, dihw, dhhw, 1024, 1024, tD);
        addbias_kernel<<<16,256,0,stream>>>(bcatA, aihb, ahhb, 4096);
        addbias_kernel<<<16,256,0,stream>>>(bcatD, dihb, dhhb, 4096);
        int tP = NP * KP;
        prep_projw_kernel<<<nblk(tP,256),256,0,stream>>>(Wproj, projw, stopw, tP);
        prep_projb_kernel<<<1,NP,0,stream>>>(bproj, stopb);
        for (int i = 0; i < 5; ++i) {
            int t = cout[i] * 5 * cpin[i];
            prep_postw_kernel<<<nblk(t,256),256,0,stream>>>(wp[i], postw[i], cout[i], cin[i], cpin[i], t);
        }
        zero_f_kernel<<<nblk(BSZ*HH,256),256,0,stream>>>(ah, BSZ*HH);
        zero_f_kernel<<<nblk(BSZ*HH,256),256,0,stream>>>(ac, BSZ*HH);
        zero_f_kernel<<<nblk(BSZ*HH,256),256,0,stream>>>(dh, BSZ*HH);
        zero_f_kernel<<<nblk(BSZ*HH,256),256,0,stream>>>(dc, BSZ*HH);
        zero_f_kernel<<<1,256,0,stream>>>(mu, BSZ*KMIX);
        zero_h_kernel<<<nblk(BSZ*KA,256),256,0,stream>>>(xcatA, BSZ*KA);
        zero_h_kernel<<<nblk(BSZ*KD,256),256,0,stream>>>(xcatD, BSZ*KD);
        zero_h_kernel<<<nblk(BSZ*KP,256),256,0,stream>>>(xcatP, BSZ*KP);
        int tact = BSZ * LPAD * 512;
        zero_h_kernel<<<nblk(tact,256),256,0,stream>>>(actB, tact);
        zero_h_kernel<<<nblk(tact,256),256,0,stream>>>(actC, tact);
    }

    // ---- 400 sequential decoder steps ----
    const int jobsLstm = 2 * (4096 / 16);   // 512 waves
    const int jobsProj = 2 * (NP / 16);     // 22 waves
    for (int t = 0; t < NSTEP; ++t) {
        attn_step_kernel<<<BSZ,256,0,stream>>>(t, mels, enc, pw1, pb1, pw2, pb2,
                                               aw1, ab1, aw2, ab2, ah, mu,
                                               xcatA, xcatP, r4);
        // attention LSTM: gates = [ctx|pre|ah] @ [Wih;Whh]^T + (bih+bhh)
        gemm_xw_kernel<<<nblk((long long)jobsLstm*32,256),256,0,stream>>>(
            xcatA, WcatA, bcatA, gates, KA, 4096, jobsLstm);
        lstm_cell_kernel<<<nblk(BSZ*HH,256),256,0,stream>>>(
            gates, ah, ac, xcatA, KA, 768, xcatD, KD, 0);
        // decoder LSTM
        gemm_xw_kernel<<<nblk((long long)jobsLstm*32,256),256,0,stream>>>(
            xcatD, WcatD, bcatD, gates, KD, 4096, jobsLstm);
        lstm_cell_kernel<<<nblk(BSZ*HH,256),256,0,stream>>>(
            gates, dh, dc, xcatD, KD, 1024, xcatP, KP, 0);
        // projection + stop
        gemm_xw_kernel<<<nblk((long long)jobsProj*32,256),256,0,stream>>>(
            xcatP, Wproj, bproj, projout, KP, NP, jobsProj);
        write_proj_kernel<<<BSZ,192,0,stream>>>(projout, r2, r3, t);
    }

    // ---- postnet ----
    {
        int tin = BSZ * LPAD * 96;
        post_prep_input_kernel<<<nblk(tin,256),256,0,stream>>>(r2, actA, tin);
        const _Float16* ins[5]  = { actA, actB, actC, actB, actC };
        _Float16*       outs[5] = { actB, actC, actB, actC, nullptr };
        for (int i = 0; i < 5; ++i) {
            int jobs = BSZ * 50 * (cout[i] / 16);
            post_conv_kernel<<<nblk((long long)jobs*32,256),256,0,stream>>>(
                ins[i], wp[i], postg[i], postb[i],
                outs[i], (i == 4) ? r1 : nullptr, (i == 4) ? r2 : nullptr,
                cpin[i], cout[i], jobs);
        }
    }
}
